// SimpleConvGCN_30863634989163
// MI455X (gfx1250) — matrix operand order repaired and checked by reading, not compiled, not run on hardware
//
#include <hip/hip_runtime.h>
#include <hip/hip_bf16.h>

typedef __attribute__((ext_vector_type(16))) __bf16 v16bf;
typedef __attribute__((ext_vector_type(8)))  float  v8f;

#define NNODES 40000
#define FIN    256
#define FOUT2  512
#define NEDGE  320000
#define NB     4
#define NPROT  10000
#define NDRUG  640
#define GOUT   32
#define H1DIM  1028
#define ZCAT   (NDRUG + GOUT)

// ---------------- elementwise helpers ----------------

__global__ void k_fill(float* p, float v, int n) {
    int i = blockIdx.x * blockDim.x + threadIdx.x;
    if (i < n) p[i] = v;
}

__global__ void k_deg_count(const long long* __restrict__ ei, float* deg) {
    int i = blockIdx.x * blockDim.x + threadIdx.x;
    if (i < NEDGE) atomicAdd(&deg[(int)ei[NEDGE + i]], 1.0f);  // col = target
}

__global__ void k_rsqrt_ip(float* p, int n) {
    int i = blockIdx.x * blockDim.x + threadIdx.x;
    if (i < n) p[i] = rsqrtf(p[i]);
}

__global__ void k_relu_ip(float* p, int n) {
    int i = blockIdx.x * blockDim.x + threadIdx.x;
    if (i < n) p[i] = fmaxf(p[i], 0.0f);
}

// ---------------- fragment packing (f32 -> bf16 in WMMA lane layout) --------
//
// A fragment (16x32 MxK, 16-bit): lane l holds row m = l&15; element e maps to
//   k = ((l&16)>>1) + e + (e&8)   within the 32-wide k-chunk.
// B fragment (32x16 KxN, 16-bit): lane l holds col n = l&15; element e maps to
//   k = ((l&16)>>1) + e + (e&8)   within the 32-wide k-chunk.
// Packed order: [tile][kchunk][lane][e] -> each lane reads 32 contiguous bytes.

__global__ void k_pack_A(const float* __restrict__ X, __bf16* __restrict__ out,
                         int M, int K, int do_relu)
{
    int idx = blockIdx.x * blockDim.x + threadIdx.x;
    int total = M * K;
    if (idx >= total) return;
    int e    = idx & 15;
    int l    = (idx >> 4) & 31;
    int rest = idx >> 9;               // mT*KC + kc
    int KC   = K >> 5;
    int kc   = rest % KC;
    int mT   = rest / KC;
    int k    = kc * 32 + ((l & 16) >> 1) + e + (e & 8);
    int m    = mT * 16 + (l & 15);
    float v  = X[(size_t)m * K + k];
    if (do_relu) v = fmaxf(v, 0.0f);
    out[idx] = (__bf16)v;
}

__global__ void k_pack_B(const float* __restrict__ W, __bf16* __restrict__ out,
                         int K, int N)
{
    int idx = blockIdx.x * blockDim.x + threadIdx.x;
    int total = K * N;
    if (idx >= total) return;
    int e    = idx & 15;
    int l    = (idx >> 4) & 31;
    int rest = idx >> 9;               // nT*KC + kc
    int KC   = K >> 5;
    int kc   = rest % KC;
    int nT   = rest / KC;
    int k    = kc * 32 + ((l & 16) >> 1) + e + (e & 8);
    int n    = nT * 16 + (l & 15);
    out[idx] = (__bf16)W[(size_t)k * N + n];
}

// ---------------- bf16 WMMA GEMM on packed fragments ------------------------
// C[M,N] = A[M,K] @ B[K,N], f32 accumulate.
// 2x2 register tile: each wave computes a 32x32 block (4 accumulators,
// 4 fragment loads -> 4 WMMAs per k-chunk). 4 waves/block cover 64x64.
// Requires M%64==0, N%64==0, K%32==0 (true for all uses here).

__global__ __launch_bounds__(128)
void k_gemm_packed(const __bf16* __restrict__ Ap, const __bf16* __restrict__ Bp,
                   float* __restrict__ C, int M, int N, int K)
{
    const int lane = threadIdx.x & 31;
    const int wave = threadIdx.x >> 5;
    const int KC   = K >> 5;
    const int mT0  = blockIdx.y * 4 + (wave >> 1) * 2;   // 16-row tile index
    const int nT0  = blockIdx.x * 4 + (wave & 1) * 2;    // 16-col tile index
    if (mT0 * 16 >= M || nT0 * 16 >= N) return;          // wave-uniform

    const __bf16* pa0 = Ap + ((size_t)(mT0    ) * KC * 32 + lane) * 16;
    const __bf16* pa1 = Ap + ((size_t)(mT0 + 1) * KC * 32 + lane) * 16;
    const __bf16* pb0 = Bp + ((size_t)(nT0    ) * KC * 32 + lane) * 16;
    const __bf16* pb1 = Bp + ((size_t)(nT0 + 1) * KC * 32 + lane) * 16;

    v8f acc00 = {}, acc01 = {}, acc10 = {}, acc11 = {};
    for (int kc = 0; kc < KC; ++kc) {
        size_t o = (size_t)kc * 512;
        v16bf a0 = *(const v16bf*)(pa0 + o);
        v16bf a1 = *(const v16bf*)(pa1 + o);
        v16bf b0 = *(const v16bf*)(pb0 + o);
        v16bf b1 = *(const v16bf*)(pb1 + o);
        acc00 = __builtin_amdgcn_wmma_f32_16x16x32_bf16(false, a0, false, b0, (short)0, acc00, false, false);
        acc01 = __builtin_amdgcn_wmma_f32_16x16x32_bf16(false, a0, false, b1, (short)0, acc01, false, false);
        acc10 = __builtin_amdgcn_wmma_f32_16x16x32_bf16(false, a1, false, b0, (short)0, acc10, false, false);
        acc11 = __builtin_amdgcn_wmma_f32_16x16x32_bf16(false, a1, false, b1, (short)0, acc11, false, false);
    }

    // D layout: VGPR j -> row tile*16 + (lane>=16 ? 8 : 0) + j, col = lane&15
    const int half = lane >> 4;
    const int l16  = lane & 15;
    {
        float* p = C + (size_t)(mT0 * 16 + half * 8) * N + nT0 * 16 + l16;
#pragma unroll
        for (int j = 0; j < 8; ++j) { p[(size_t)j * N] = acc00[j]; p[(size_t)j * N + 16] = acc01[j]; }
    }
    {
        float* p = C + (size_t)((mT0 + 1) * 16 + half * 8) * N + nT0 * 16 + l16;
#pragma unroll
        for (int j = 0; j < 8; ++j) { p[(size_t)j * N] = acc10[j]; p[(size_t)j * N + 16] = acc11[j]; }
    }
}

// ---------------- GCN aggregation ----------------

// O[node][c] = H[node][c] * dinv[node]^2 + bias[c]   (self-loop + bias)
__global__ void k_agg_init(const float* __restrict__ H, const float* __restrict__ dinv,
                           const float* __restrict__ bias, float* __restrict__ O, int C)
{
    int idx = blockIdx.x * blockDim.x + threadIdx.x;
    int total = NNODES * C;
    if (idx >= total) return;
    int node = idx / C;
    int c = idx - node * C;
    float d = dinv[node];
    O[idx] = H[idx] * d * d + bias[c];
}

// O[col] += H[row] * dinv[row]*dinv[col], 4 feats per thread
__global__ void k_edge_scatter(const long long* __restrict__ ei, const float* __restrict__ dinv,
                               const float* __restrict__ H, float* __restrict__ O, int C)
{
    int idx = blockIdx.x * blockDim.x + threadIdx.x;
    int chunks = C >> 2;
    int total = NEDGE * chunks;
    if (idx >= total) return;
    int e = idx / chunks;
    int c = (idx - e * chunks) << 2;
    int r = (int)ei[e];
    int t = (int)ei[NEDGE + e];
    float w = dinv[r] * dinv[t];
    const float* hp = H + (size_t)r * C + c;
    float* op = O + (size_t)t * C + c;
    atomicAdd(op + 0, hp[0] * w);
    atomicAdd(op + 1, hp[1] * w);
    atomicAdd(op + 2, hp[2] * w);
    atomicAdd(op + 3, hp[3] * w);
}

// ---------------- pooling & MLP head ----------------

__global__ void k_maxpool(const float* __restrict__ O, float* __restrict__ g, int C)
{
    int c = blockIdx.x;
    int b = blockIdx.y;
    float m = -3.4e38f;
    for (int i = threadIdx.x; i < NPROT; i += blockDim.x)
        m = fmaxf(m, O[((size_t)b * NPROT + i) * C + c]);
    __shared__ float s[256];
    s[threadIdx.x] = m;
    __syncthreads();
    for (int off = 128; off > 0; off >>= 1) {
        if ((int)threadIdx.x < off) s[threadIdx.x] = fmaxf(s[threadIdx.x], s[threadIdx.x + off]);
        __syncthreads();
    }
    if (threadIdx.x == 0) g[b * C + c] = s[0];
}

// Y[M,Nout] = act(X[M,K] @ W[K,Nout] + bias); act 0=id 1=relu 2=sigmoid
__global__ void k_fc(const float* __restrict__ X, const float* __restrict__ W,
                     const float* __restrict__ bias, float* __restrict__ Y,
                     int M, int K, int Nout, int act)
{
    int idx = blockIdx.x * blockDim.x + threadIdx.x;
    if (idx >= M * Nout) return;
    int m = idx / Nout;
    int n = idx - m * Nout;
    float acc = bias[n];
    for (int k = 0; k < K; ++k) acc += X[m * K + k] * W[k * Nout + n];
    if (act == 1)      acc = fmaxf(acc, 0.0f);
    else if (act == 2) acc = 1.0f / (1.0f + expf(-acc));
    Y[idx] = acc;
}

// z = concat([g2 (4x32), DDI (4x640)], dim=1) -> 4x672
__global__ void k_concat(const float* __restrict__ g2, const float* __restrict__ ddi,
                         float* __restrict__ z)
{
    int idx = blockIdx.x * blockDim.x + threadIdx.x;
    if (idx >= NB * ZCAT) return;
    int b = idx / ZCAT;
    int j = idx - b * ZCAT;
    z[idx] = (j < GOUT) ? g2[b * GOUT + j] : ddi[b * NDRUG + (j - GOUT)];
}

// ---------------- host launch ----------------

static inline size_t align256(size_t x) { return (x + 255) & ~(size_t)255; }

extern "C" void kernel_launch(void* const* d_in, const int* in_sizes, int n_in,
                              void* d_out, int out_size, void* d_ws, size_t ws_size,
                              hipStream_t stream)
{
    (void)in_sizes; (void)n_in; (void)out_size; (void)ws_size;

    const float*      x   = (const float*)d_in[0];
    const long long*  ei  = (const long long*)d_in[1];
    const float*      ddi = (const float*)d_in[2];
    const float*      W1  = (const float*)d_in[3];
    const float*      b1  = (const float*)d_in[4];
    const float*      W2  = (const float*)d_in[5];
    const float*      b2  = (const float*)d_in[6];
    const float*      Wg1 = (const float*)d_in[7];
    const float*      bg1 = (const float*)d_in[8];
    const float*      Wg2 = (const float*)d_in[9];
    const float*      bg2 = (const float*)d_in[10];
    const float*      Wf1 = (const float*)d_in[11];
    const float*      bf1 = (const float*)d_in[12];
    const float*      Wf2 = (const float*)d_in[13];
    const float*      bf2 = (const float*)d_in[14];
    const float*      Wf3 = (const float*)d_in[15];
    const float*      bf3 = (const float*)d_in[16];
    float*            out = (float*)d_out;

    // workspace arena
    char* ws = (char*)d_ws;
    size_t off = 0;
    float*  dinv = (float*)(ws + off);  off = align256(off + (size_t)NNODES * 4);
    __bf16* Ap   = (__bf16*)(ws + off); off = align256(off + (size_t)NNODES * FIN * 2);
    __bf16* Bp   = (__bf16*)(ws + off); off = align256(off + (size_t)FIN * FOUT2 * 2);
    float*  Hb   = (float*)(ws + off);  off = align256(off + (size_t)NNODES * FOUT2 * 4);
    float*  Ob   = (float*)(ws + off);  off = align256(off + (size_t)NNODES * FOUT2 * 4);
    float*  g    = (float*)(ws + off);  off = align256(off + (size_t)NB * FOUT2 * 4);
    float*  g1   = (float*)(ws + off);  off = align256(off + (size_t)NB * H1DIM * 4);
    float*  g2   = (float*)(ws + off);  off = align256(off + (size_t)NB * GOUT * 4);
    float*  z    = (float*)(ws + off);  off = align256(off + (size_t)NB * ZCAT * 4);
    float*  z1   = (float*)(ws + off);  off = align256(off + (size_t)NB * 64 * 4);
    float*  z2   = (float*)(ws + off);  off = align256(off + (size_t)NB * 16 * 4);

    const int T = 256;
    #define GRID(n) (((n) + T - 1) / T)

    // degrees: deg = 1 (self loop) + count on target, then rsqrt in place
    k_fill<<<GRID(NNODES), T, 0, stream>>>(dinv, 1.0f, NNODES);
    k_deg_count<<<GRID(NEDGE), T, 0, stream>>>(ei, dinv);
    k_rsqrt_ip<<<GRID(NNODES), T, 0, stream>>>(dinv, NNODES);

    // ---- layer 1: H = x @ W1 (bf16 WMMA on packed fragments) ----
    k_pack_A<<<GRID(NNODES * FIN), T, 0, stream>>>(x, Ap, NNODES, FIN, 0);
    k_pack_B<<<GRID(FIN * FIN), T, 0, stream>>>(W1, Bp, FIN, FIN);
    {
        dim3 grid(FIN / 64, NNODES / 64);
        k_gemm_packed<<<grid, 128, 0, stream>>>(Ap, Bp, Hb, NNODES, FIN, FIN);
    }
    k_agg_init<<<GRID(NNODES * FIN), T, 0, stream>>>(Hb, dinv, b1, Ob, FIN);
    k_edge_scatter<<<GRID(NEDGE * (FIN / 4)), T, 0, stream>>>(ei, dinv, Hb, Ob, FIN);

    // ---- layer 2: H2 = relu(O1) @ W2 ----
    // relu fused into the A-fragment repack (reuses Ap)
    k_pack_A<<<GRID(NNODES * FIN), T, 0, stream>>>(Ob, Ap, NNODES, FIN, 1);
    k_pack_B<<<GRID(FIN * FOUT2), T, 0, stream>>>(W2, Bp, FIN, FOUT2);
    {
        dim3 grid(FOUT2 / 64, NNODES / 64);
        k_gemm_packed<<<grid, 128, 0, stream>>>(Ap, Bp, Hb, NNODES, FOUT2, FIN);
    }
    k_agg_init<<<GRID(NNODES * FOUT2), T, 0, stream>>>(Hb, dinv, b2, Ob, FOUT2);
    k_edge_scatter<<<GRID(NEDGE * (FOUT2 / 4)), T, 0, stream>>>(ei, dinv, Hb, Ob, FOUT2);
    k_relu_ip<<<GRID(NNODES * FOUT2), T, 0, stream>>>(Ob, NNODES * FOUT2);

    // ---- global max pool per graph ----
    {
        dim3 grid(FOUT2, NB);
        k_maxpool<<<grid, 256, 0, stream>>>(Ob, g, FOUT2);
    }

    // ---- MLP head ----
    k_fc<<<GRID(NB * H1DIM), T, 0, stream>>>(g, Wg1, bg1, g1, NB, FOUT2, H1DIM, 1);
    k_fc<<<GRID(NB * GOUT),  T, 0, stream>>>(g1, Wg2, bg2, g2, NB, H1DIM, GOUT, 0);
    k_concat<<<GRID(NB * ZCAT), T, 0, stream>>>(g2, ddi, z);
    k_fc<<<GRID(NB * 64), T, 0, stream>>>(z,  Wf1, bf1, z1, NB, ZCAT, 64, 1);
    k_fc<<<GRID(NB * 16), T, 0, stream>>>(z1, Wf2, bf2, z2, NB, 64, 16, 1);
    k_fc<<<GRID(NB * 1),  T, 0, stream>>>(z2, Wf3, bf3, out, NB, 16, 1, 2);

    #undef GRID
}
